// FuturePredictionODE_50594714747062
// MI455X (gfx1250) — matrix operand order repaired and checked by reading, not compiled, not run on hardware
//
#include <hip/hip_runtime.h>
#include <hip/hip_bf16.h>
#include <math.h>

typedef __attribute__((ext_vector_type(16))) _Float16 v16h;
typedef __attribute__((ext_vector_type(8)))  float    v8f;

#define BB 2
#define CC 64
#define HH 80
#define WW2 80
#define HWP (HH*WW2)            // 6400
#define TT ((long)CC*HWP)       // 409600 floats per (C,H,W) image
#define SS 6
#define DT_C 0.1f

// activation codes
#define ACT_NONE 0
#define ACT_RELU 1
#define ACT_SIG  2
#define ACT_TANH 3

// ---------------------------------------------------------------------------
// Implicit-GEMM conv via v_wmma_f32_16x16x32_f16.
// Block = 128 threads (4 waves) computing a 64(cout) x 64(pixel) macro-tile.
// Per 32-wide K chunk: stage 4 A-tiles (one per wave's cout range, row-major
// [m][k], contiguous K per lane half) and 4 B-tiles (transposed per 16-pixel tile
// so each lane's 16 halves are contiguous), then issue 16 WMMAs (A reused
// across the 4 N-tiles, B shared across the 4 cout waves).
// Input may be a virtual concat of two tensors split at channel c_split.
// Weight layout OIHW with row width wcin; ci_off selects a channel slice
// (used for the 640-ch ASPP projection done as accumulating 128-ch convs).
// ---------------------------------------------------------------------------
template<int KS>
__global__ __launch_bounds__(128)
void conv_wmma(const float* __restrict__ in0, const float* __restrict__ in1,
               int c_split, int cin, long istr0, long istr1,
               const float* __restrict__ wgt, int wcin, int ci_off,
               const float* __restrict__ bias,
               float* __restrict__ out, long ostr,
               int dil, int act, int accum)
{
    constexpr int KK = KS * KS;
    __shared__ alignas(32) _Float16 lA[4 * 512]; // 4 cout tiles, row-major [m][k]
    __shared__ alignas(32) _Float16 lB[4 * 512]; // 4 pixel tiles, transposed

    const int tid  = threadIdx.x;
    const int wave = tid >> 5;
    const int lane = tid & 31;
    const int l16  = lane & 15;
    const int hi   = lane >> 4;

    const int pixBase = blockIdx.x * 64;
    const int coB     = blockIdx.y * 64;
    const int img     = blockIdx.z;

    const float* src0 = in0 + (long)img * istr0;
    const float* src1 = in1 + (long)img * istr1;

    const int Ktot = cin * KK;
    const int pad  = KS >> 1;

    v8f c0 = {}, c1 = {}, c2 = {}, c3 = {};

    for (int kb = 0; kb < Ktot; kb += 32) {
        // ---- stage A: 64m x 32k = 2048 halves, 16/thread, coalesced in k ----
        #pragma unroll
        for (int i = 0; i < 16; ++i) {
            int e = i * 128 + tid;
            int m = e >> 5, k = e & 31;
            int kg = kb + k;
            int ci = kg / KK, r = kg - ci * KK;
            float wv = wgt[((long)(coB + m) * wcin + (ci_off + ci)) * KK + r];
            // A-operand per-lane order is contiguous K from hi*16:
            // lane half j: lanes0-15 -> K {0-15}; lanes16-31 -> K {16-31}
            int koff = k;
            lA[(m >> 4) * 512 + (m & 15) * 32 + koff] = (_Float16)wv;
        }
        // ---- stage B: 32k x 64n = 2048 halves, transposed per 16-pix tile ----
        #pragma unroll
        for (int i = 0; i < 16; ++i) {
            int e = i * 128 + tid;
            int k = e >> 6, n = e & 63;
            int kg = kb + k;
            int ci = kg / KK, r = kg - ci * KK;
            int ky = r / KS, kx = r - ky * KS;
            int pix = pixBase + n;
            int y = pix / WW2, x = pix - y * WW2;
            int iy = y + dil * (ky - pad);
            int ix = x + dil * (kx - pad);
            float v = 0.f;
            if (iy >= 0 && iy < HH && ix >= 0 && ix < WW2) {
                const float* s = (ci < c_split)
                    ? (src0 + (long)ci * HWP)
                    : (src1 + (long)(ci - c_split) * HWP);
                v = s[iy * WW2 + ix];
            }
            // B-operand per-lane order is contiguous K from hi*16
            lB[(n >> 4) * 512 + (n & 15) * 32 + k] = (_Float16)v;
        }
        __syncthreads();

        // ---- 32-byte vector operand reads + 4 WMMAs per wave ----
        v16h a  = *(const v16h*)&lA[wave * 512 + l16 * 32 + hi * 16];
        v16h b0 = *(const v16h*)&lB[0 * 512   + l16 * 32 + hi * 16];
        v16h b1 = *(const v16h*)&lB[1 * 512   + l16 * 32 + hi * 16];
        v16h b2 = *(const v16h*)&lB[2 * 512   + l16 * 32 + hi * 16];
        v16h b3 = *(const v16h*)&lB[3 * 512   + l16 * 32 + hi * 16];
        c0 = __builtin_amdgcn_wmma_f32_16x16x32_f16(false, a, false, b0, (short)0, c0, false, false);
        c1 = __builtin_amdgcn_wmma_f32_16x16x32_f16(false, a, false, b1, (short)0, c1, false, false);
        c2 = __builtin_amdgcn_wmma_f32_16x16x32_f16(false, a, false, b2, (short)0, c2, false, false);
        c3 = __builtin_amdgcn_wmma_f32_16x16x32_f16(false, a, false, b3, (short)0, c3, false, false);
        __syncthreads();
    }

    // ---- epilogue: C layout (VGPR v -> M = v + 8*hi, N = l16) -> NCHW ----
    float* dst = out + (long)img * ostr;
    const int mOff = hi * 8;
    #pragma unroll
    for (int nt = 0; nt < 4; ++nt) {
        v8f cc = (nt == 0) ? c0 : (nt == 1) ? c1 : (nt == 2) ? c2 : c3;
        int pix = pixBase + nt * 16 + l16;
        #pragma unroll
        for (int v = 0; v < 8; ++v) {
            int co = coB + wave * 16 + mOff + v;
            float val = cc[v];
            if (bias) val += bias[co];
            long oidx = (long)co * HWP + pix;
            if (accum) val += dst[oidx];
            if      (act == ACT_RELU) val = fmaxf(val, 0.f);
            else if (act == ACT_SIG)  val = 1.f / (1.f + __expf(-val));
            else if (act == ACT_TANH) val = tanhf(val);
            dst[oidx] = val;
        }
    }
}

// ---------------------------------------------------------------------------
// Elementwise / utility kernels
// ---------------------------------------------------------------------------
__global__ void ew_mul(const float* a, const float* b, float* o, long n) {
    long i = (long)blockIdx.x * blockDim.x + threadIdx.x;
    if (i < n) o[i] = a[i] * b[i];
}
// out = (1-g)*x0 + g*x1
__global__ void ew_gate_mix(const float* g, const float* x0, const float* x1,
                            float* o, long n) {
    long i = (long)blockIdx.x * blockDim.x + threadIdx.x;
    if (i < n) { float gg = g[i]; o[i] = (1.f - gg) * x0[i] + gg * x1[i]; }
}
// h += DT*(1-z)*(gv-h)
__global__ void ew_euler(const float* z, const float* gv, float* h, long n) {
    long i = (long)blockIdx.x * blockDim.x + threadIdx.x;
    if (i < n) { float hv = h[i]; h[i] = hv + DT_C * (1.f - z[i]) * (gv[i] - hv); }
}
__global__ void ew_relu(float* a, long n) {
    long i = (long)blockIdx.x * blockDim.x + threadIdx.x;
    if (i < n) a[i] = fmaxf(a[i], 0.f);
}
__global__ void ew_zero(float* a, long n) {
    long i = (long)blockIdx.x * blockDim.x + threadIdx.x;
    if (i < n) a[i] = 0.f;
}
__global__ void copy_imgs(float* dst, long dstr, const float* src, long sstr,
                          long n) {
    long i = (long)blockIdx.x * blockDim.x + threadIdx.x;
    int img = blockIdx.y;
    if (i < n) dst[(long)img * dstr + i] = src[(long)img * sstr + i];
}
__global__ void bcast_init(const float* basev, float* dst, long npix, long n) {
    long i = (long)blockIdx.x * blockDim.x + threadIdx.x;
    if (i < n) dst[i] = basev[i / npix];
}
__global__ void compute_order(const float* ct, const float* lt, int* ord) {
    int b = threadIdx.x;
    if (b >= BB) return;
    float t[6]; int idx[6];
    for (int i = 0; i < 3; ++i) { t[i] = ct[b*3+i]; idx[i] = i; }
    for (int i = 0; i < 3; ++i) { t[3+i] = lt[b*3+i]; idx[3+i] = 3+i; }
    for (int i = 0; i < 6; ++i)
        for (int j = i + 1; j < 6; ++j)
            if (t[j] < t[i]) {
                float tv = t[i]; t[i] = t[j]; t[j] = tv;
                int ti = idx[i]; idx[i] = idx[j]; idx[j] = ti;
            }
    for (int i = 0; i < 6; ++i) ord[b*6+i] = idx[i];
}
__global__ void gather_obs(const float* cam, const float* lid, const int* ord,
                           float* dst, long total) {
    long i = (long)blockIdx.x * blockDim.x + threadIdx.x;
    if (i >= total) return;
    long per = 6 * TT;
    int b = (int)(i / per);
    long rem = i - (long)b * per;
    int j = (int)(rem / TT);
    long off = rem - (long)j * TT;
    int o = ord[b*6+j];
    dst[i] = (o < 3) ? cam[((long)b*3 + o) * TT + off]
                     : lid[((long)b*3 + (o-3)) * TT + off];
}
// per-batch sum of squared differences, accumulated into aux[b]
__global__ void mse_accum(const float* a, long astr, const float* bo, long bstr,
                          float* aux, long n) {
    int b = blockIdx.y;
    __shared__ float sm[256];
    float s = 0.f;
    for (long i = (long)blockIdx.x * blockDim.x + threadIdx.x; i < n;
         i += (long)gridDim.x * blockDim.x) {
        float d = a[(long)b*astr + i] - bo[(long)b*bstr + i];
        s += d * d;
    }
    sm[threadIdx.x] = s; __syncthreads();
    for (int o = 128; o > 0; o >>= 1) {
        if (threadIdx.x < o) sm[threadIdx.x] += sm[threadIdx.x + o];
        __syncthreads();
    }
    if (threadIdx.x == 0) atomicAdd(&aux[b], sm[0]);
}
// one image: 64 channels x 6400 -> pool[c]
__global__ void avgpool64(const float* x, float* pool) {
    int c = blockIdx.x;
    __shared__ float sm[256];
    float s = 0.f;
    for (int i = threadIdx.x; i < HWP; i += 256) s += x[(long)c * HWP + i];
    sm[threadIdx.x] = s; __syncthreads();
    for (int o = 128; o > 0; o >>= 1) {
        if (threadIdx.x < o) sm[threadIdx.x] += sm[threadIdx.x + o];
        __syncthreads();
    }
    if (threadIdx.x == 0) pool[c] = sm[0] / (float)HWP;
}
// small dense: out[co] = act(w[co*stride+off .. +cin] . inv + bias[co]), 128 rows
__global__ void fc128(const float* w, const float* bias, const float* inv,
                      float* outv, int cin, int w_off, int w_stride, int relu) {
    int co = threadIdx.x;
    if (co >= 128) return;
    float s = bias ? bias[co] : 0.f;
    for (int i = 0; i < cin; ++i) s += w[(long)co * w_stride + w_off + i] * inv[i];
    if (relu) s = fmaxf(s, 0.f);
    outv[co] = s;
}
__global__ void finalize_aux(const float* aux, float* o, float scale) {
    o[0] = aux[BB-1] * scale;
}

// ---------------------------------------------------------------------------
extern "C" void kernel_launch(void* const* d_in, const int* in_sizes, int n_in,
                              void* d_out, int out_size, void* d_ws, size_t ws_size,
                              hipStream_t stream)
{
    const float* fut = (const float*)d_in[0];
    const float* cam = (const float*)d_in[1];
    const float* lid = (const float*)d_in[2];
    const float* cts = (const float*)d_in[3];
    const float* lts = (const float*)d_in[4];
    // d_in[5]: target_timestamp (unused by reference forward)
    // params flattened in sorted-key pytree order:
    const float* b0_b = (const float*)d_in[6];
    const float* b0_w = (const float*)d_in[7];
    const float* g_br[2] = {(const float*)d_in[8],  (const float*)d_in[13]};
    const float* g_bu[2] = {(const float*)d_in[9],  (const float*)d_in[14]};
    const float* g_wc[2] = {(const float*)d_in[10], (const float*)d_in[15]};
    const float* g_wr[2] = {(const float*)d_in[11], (const float*)d_in[16]};
    const float* g_wu[2] = {(const float*)d_in[12], (const float*)d_in[17]};
    const float* a_b[4] = {(const float*)d_in[18], (const float*)d_in[20],
                           (const float*)d_in[22], (const float*)d_in[24]};
    const float* a_w[4] = {(const float*)d_in[19], (const float*)d_in[21],
                           (const float*)d_in[23], (const float*)d_in[25]};
    const float* apb = (const float*)d_in[26];
    const float* apw = (const float*)d_in[27];
    const float* h1b = (const float*)d_in[28];
    const float* h1w = (const float*)d_in[29];
    const float* h3b = (const float*)d_in[30];
    const float* h3w = (const float*)d_in[31];
    const float* projb = (const float*)d_in[32];
    const float* projw = (const float*)d_in[33];
    const float* jbn = (const float*)d_in[34];
    const float* jbr = (const float*)d_in[35];
    const float* jbz = (const float*)d_in[36];
    const float* jn_w = (const float*)d_in[37];
    const float* jr_w = (const float*)d_in[38];
    const float* jz_w = (const float*)d_in[39];
    const float* o_bg = (const float*)d_in[40];
    const float* o_br = (const float*)d_in[41];
    const float* o_bz = (const float*)d_in[42];
    const float* o_wg = (const float*)d_in[43];
    const float* o_wr = (const float*)d_in[44];
    const float* o_wz = (const float*)d_in[45];

    // ---- workspace carve ----
    float* ws = (float*)d_ws;
    long off = 0;
    float* obs = ws + off;     off += (long)BB * 6 * TT;   // sorted observations
    float* hB  = ws + off;     off += BB * TT;             // latent state
    float* rB  = ws + off;     off += BB * TT;             // gate buf (r / u)
    float* zB  = ws + off;     off += BB * TT;             // gate buf (z / r2)
    float* gB  = ws + off;     off += BB * TT;             // gate buf (g / c)
    float* tmp = ws + off;     off += BB * TT;             // r*h
    float* hid0 = ws + off;    off += BB * TT;             // preds[:,0]
    float* hst  = ws + off;    off += BB * TT;             // running GRU state
    float* xA  = ws + off;     off += (long)BB * 6 * TT;   // sequence buffer
    float* brb = ws + off;     off += (long)128 * HWP;     // ASPP branch / h3 out
    float* prj = ws + off;     off += (long)128 * HWP;     // ASPP projection
    float* pool = ws + off;    off += 64;
    float* gpv  = ws + off;    off += 128;
    float* pbase = ws + off;   off += 128;
    float* aux  = ws + off;    off += 8;
    int*   ordp = (int*)(ws + off); off += 16;

    float* xout = (float*)d_out;                 // (B,S,C,H,W) region
    const long XSIZE = (long)BB * SS * TT;

    const long BT = (long)BB * TT;
    const int EWB = 256;
    const long ewg = (BT + EWB - 1) / EWB;

    auto conv = [&](const float* i0, long s0, const float* i1, long s1,
                    int csplit, int cin, const float* w, int wcin, int cioff,
                    const float* bias, float* outp, long ostr, int nimg,
                    int cout, int ks, int dil, int act, int accum) {
        dim3 g(HWP / 64, cout / 64, nimg);
        if (ks == 1)
            conv_wmma<1><<<g, 128, 0, stream>>>(i0, i1, csplit, cin, s0, s1,
                                                w, wcin, cioff, bias, outp, ostr,
                                                dil, act, accum);
        else
            conv_wmma<3><<<g, 128, 0, stream>>>(i0, i1, csplit, cin, s0, s1,
                                                w, wcin, cioff, bias, outp, ostr,
                                                dil, act, accum);
    };
    auto jump = [&](const float* obk) {   // h = jump(h, ob)
        conv(obk, 6*TT, hB, TT, CC, 2*CC, jr_w, 2*CC, 0, jbr, rB, TT, BB, CC, 3, 1, ACT_SIG, 0);
        conv(obk, 6*TT, hB, TT, CC, 2*CC, jz_w, 2*CC, 0, jbz, zB, TT, BB, CC, 3, 1, ACT_SIG, 0);
        ew_mul<<<ewg, EWB, 0, stream>>>(rB, hB, tmp, BT);
        conv(obk, 6*TT, tmp, TT, CC, 2*CC, jn_w, 2*CC, 0, jbn, gB, TT, BB, CC, 3, 1, ACT_TANH, 0);
        ew_gate_mix<<<ewg, EWB, 0, stream>>>(zB, gB, hB, hB, BT);  // (1-z)*n + z*h
    };
    auto euler = [&]() {
        conv(hB, TT, hB, TT, CC, CC, o_wr, CC, 0, o_br, rB, TT, BB, CC, 3, 1, ACT_SIG, 0);
        conv(hB, TT, hB, TT, CC, CC, o_wz, CC, 0, o_bz, zB, TT, BB, CC, 3, 1, ACT_SIG, 0);
        ew_mul<<<ewg, EWB, 0, stream>>>(rB, hB, tmp, BT);
        conv(tmp, TT, tmp, TT, CC, CC, o_wg, CC, 0, o_bg, gB, TT, BB, CC, 3, 1, ACT_TANH, 0);
        ew_euler<<<ewg, EWB, 0, stream>>>(zB, gB, hB, BT);
    };

    // ---- 1. sort observation streams by timestamp, gather ----
    compute_order<<<1, 32, 0, stream>>>(cts, lts, ordp);
    {
        long total = (long)BB * 6 * TT;
        gather_obs<<<(unsigned)((total + EWB - 1) / EWB), EWB, 0, stream>>>(cam, lid, ordp, obs, total);
    }
    ew_zero<<<1, 32, 0, stream>>>(aux, 8);
    hipMemcpyAsync(hB, fut, BT * sizeof(float), hipMemcpyDeviceToDevice, stream);

    // ---- 2. ODE filter over observations ----
    mse_accum<<<dim3(64, BB), 256, 0, stream>>>(hB, TT, obs + 0, 6*TT, aux, TT);
    jump(obs + 0);
    for (int k = 1; k < 6; ++k) {
        euler(); euler();
        mse_accum<<<dim3(64, BB), 256, 0, stream>>>(hB, TT, obs + (long)k*TT, 6*TT, aux, TT);
        jump(obs + (long)k*TT);
    }

    // ---- 3. future rollout: preds -> xA ----
    for (int t = 0; t < SS; ++t) {
        euler(); euler();
        copy_imgs<<<dim3((unsigned)((TT + EWB - 1) / EWB), BB), EWB, 0, stream>>>(
            xA + (long)t*TT, 6*TT, hB, TT, TT);
    }
    copy_imgs<<<dim3((unsigned)((TT + EWB - 1) / EWB), BB), EWB, 0, stream>>>(
        hid0, TT, xA, 6*TT, TT);

    // ---- 4. GRU layer 0 (in-place on xA), then block0 -> d_out ----
    hipMemcpyAsync(hst, hid0, BT * sizeof(float), hipMemcpyDeviceToDevice, stream);
    for (int t = 0; t < SS; ++t) {
        const float* xt = xA + (long)t*TT;
        conv(xt, 6*TT, hst, TT, CC, 2*CC, g_wu[0], 2*CC, 0, g_bu[0], rB, TT, BB, CC, 3, 1, ACT_SIG, 0);
        conv(xt, 6*TT, hst, TT, CC, 2*CC, g_wr[0], 2*CC, 0, g_br[0], zB, TT, BB, CC, 3, 1, ACT_SIG, 0);
        ew_mul<<<ewg, EWB, 0, stream>>>(zB, hst, tmp, BT);
        conv(xt, 6*TT, tmp, TT, CC, 2*CC, g_wc[0], 2*CC, 0, nullptr, gB, TT, BB, CC, 3, 1, ACT_TANH, 0);
        ew_gate_mix<<<ewg, EWB, 0, stream>>>(rB, hst, gB, hst, BT);   // (1-u)*h + u*c
        copy_imgs<<<dim3((unsigned)((TT + EWB - 1) / EWB), BB), EWB, 0, stream>>>(
            xA + (long)t*TT, 6*TT, hst, TT, TT);
    }
    // block0: relu(conv3x3) over all B*S images, xA -> d_out
    conv(xA, TT, xA, TT, CC, CC, b0_w, CC, 0, b0_b, xout, TT, BB*SS, CC, 3, 1, ACT_RELU, 0);

    // ---- 5. GRU layer 1 (in-place on d_out) ----
    hipMemcpyAsync(hst, hid0, BT * sizeof(float), hipMemcpyDeviceToDevice, stream);
    for (int t = 0; t < SS; ++t) {
        const float* xt = xout + (long)t*TT;
        conv(xt, 6*TT, hst, TT, CC, 2*CC, g_wu[1], 2*CC, 0, g_bu[1], rB, TT, BB, CC, 3, 1, ACT_SIG, 0);
        conv(xt, 6*TT, hst, TT, CC, 2*CC, g_wr[1], 2*CC, 0, g_br[1], zB, TT, BB, CC, 3, 1, ACT_SIG, 0);
        ew_mul<<<ewg, EWB, 0, stream>>>(zB, hst, tmp, BT);
        conv(xt, 6*TT, tmp, TT, CC, 2*CC, g_wc[1], 2*CC, 0, nullptr, gB, TT, BB, CC, 3, 1, ACT_TANH, 0);
        ew_gate_mix<<<ewg, EWB, 0, stream>>>(rB, hst, gB, hst, BT);
        copy_imgs<<<dim3((unsigned)((TT + EWB - 1) / EWB), BB), EWB, 0, stream>>>(
            xout + (long)t*TT, 6*TT, hst, TT, TT);
    }

    // ---- 6. DeepLab ASPP head, per image, in place on d_out ----
    const int rates[3] = {12, 24, 36};
    const long PRJN = (long)128 * HWP;
    const long prjg = (PRJN + EWB - 1) / EWB;
    for (int img = 0; img < BB * SS; ++img) {
        float* xi = xout + (long)img * TT;
        // pooled branch -> constant per-channel projection contribution
        avgpool64<<<64, 256, 0, stream>>>(xi, pool);
        fc128<<<1, 128, 0, stream>>>(apw, apb, pool, gpv, 64, 0, 64, 1);
        fc128<<<1, 128, 0, stream>>>(projw, projb, gpv, pbase, 128, 512, 640, 0);
        bcast_init<<<(unsigned)prjg, EWB, 0, stream>>>(pbase, prj, HWP, PRJN);
        // branch 0: 1x1 ; branches 1..3: 3x3 dilated
        for (int bi = 0; bi < 4; ++bi) {
            int ks = (bi == 0) ? 1 : 3;
            int dl = (bi == 0) ? 1 : rates[bi - 1];
            conv(xi, 0, xi, 0, CC, CC, a_w[bi], CC, 0, a_b[bi], brb, 0, 1, 128, ks, dl, ACT_RELU, 0);
            conv(brb, 0, brb, 0, 128, 128, projw, 640, bi * 128, nullptr, prj, 0, 1, 128, 1, 1, ACT_NONE, 1);
        }
        ew_relu<<<(unsigned)prjg, EWB, 0, stream>>>(prj, PRJN);
        conv(prj, 0, prj, 0, 128, 128, h3w, 128, 0, h3b, brb, 0, 1, 128, 3, 1, ACT_RELU, 0);
        conv(brb, 0, brb, 0, 128, 128, h1w, 128, 0, h1b, xi, 0, 1, CC, 1, 1, ACT_NONE, 0);
    }

    // ---- 7. aux scalar: aux[-1] (sum of per-step means) ----
    finalize_aux<<<1, 1, 0, stream>>>(aux, xout + XSIZE, 1.f / (float)TT);
}
